// MultiHeadSelfAttention_70600672412212
// MI455X (gfx1250) — matrix operand order repaired and checked by reading, not compiled, not run on hardware
//
#include <hip/hip_runtime.h>

// ---------------- CDNA5 WMMA vector types ----------------
typedef __attribute__((ext_vector_type(16))) __bf16 v16bf;
typedef __attribute__((ext_vector_type(8)))  __bf16 v8bf;
typedef __attribute__((ext_vector_type(8)))  float  v8f;

__device__ __forceinline__ v16bf cat8(v8bf lo, v8bf hi) {
  return __builtin_shufflevector(lo, hi, 0,1,2,3,4,5,6,7,8,9,10,11,12,13,14,15);
}

__device__ __forceinline__ v8f wmma_bf16(v16bf a, v16bf b, v8f c) {
  return __builtin_amdgcn_wmma_f32_16x16x32_bf16(false, a, false, b, (short)0, c, false, false);
}

#define TB   2048
#define CB   1024
#define HB   16
#define DB   64

// ---------------- elementwise f32 -> bf16 ----------------
__global__ void k_cvt_bf16(const float* __restrict__ in, __bf16* __restrict__ out, int n) {
  int i = blockIdx.x * blockDim.x + threadIdx.x;
  if (i < n) out[i] = (__bf16)in[i];
}

// ---------------- transpose f32[K][N] -> bf16[N][K] ----------------
__global__ void k_transpose_bf16(const float* __restrict__ W, __bf16* __restrict__ Wt) {
  int i = blockIdx.x * blockDim.x + threadIdx.x;            // i = k*1024 + n
  if (i < CB * CB) {
    int k = i >> 10, n = i & 1023;
    Wt[(size_t)n * CB + k] = (__bf16)W[i];
  }
}

// ---------------- register-blocked GEMM: 32x64 per wave, bf16 output ----------------
// A: xb[M,K] bf16 row-major; Bt: weight [N,K] bf16 (pre-transposed)
// OUTMODE 0: out[b,h,t,d]   (Q/K head layout)
// OUTMODE 1: out[b,h,d,t]   (V head-transposed, so flash V B-fragments are contiguous)
template <int OUTMODE>
__global__ void k_gemm_bf16out(const __bf16* __restrict__ A, const __bf16* __restrict__ Bt,
                               const float* __restrict__ bias, __bf16* __restrict__ out) {
  const int K = CB;
  int w    = blockIdx.x * (blockDim.x >> 5) + (threadIdx.x >> 5);
  int lane = threadIdx.x & 31, half = lane >> 4, lc = lane & 15;
  int mtile = w >> 4, ntile = w & 15;                       // M/32 = 128, N/64 = 16
  int m0 = mtile << 5, n0 = ntile << 6;

  const __bf16* arow0 = A + (size_t)(m0 + lc) * K;
  const __bf16* arow1 = A + (size_t)(m0 + 16 + lc) * K;
  const __bf16* brow[4];
#pragma unroll
  for (int j = 0; j < 4; ++j)
    brow[j] = Bt + (size_t)(n0 + 16 * j + lc) * K + half * 16;

  v8f acc[2][4] = {};
  for (int kk = 0; kk < K; kk += 32) {
    v16bf a0 = cat8(*(const v8bf*)(arow0 + kk + half * 16),
                    *(const v8bf*)(arow0 + kk + 8 + half * 16));
    v16bf a1 = cat8(*(const v8bf*)(arow1 + kk + half * 16),
                    *(const v8bf*)(arow1 + kk + 8 + half * 16));
    v16bf b[4];
#pragma unroll
    for (int j = 0; j < 4; ++j) b[j] = *(const v16bf*)(brow[j] + kk);
#pragma unroll
    for (int j = 0; j < 4; ++j) {
      acc[0][j] = wmma_bf16(a0, b[j], acc[0][j]);
      acc[1][j] = wmma_bf16(a1, b[j], acc[1][j]);
    }
  }
#pragma unroll
  for (int i = 0; i < 2; ++i) {
#pragma unroll
    for (int j = 0; j < 4; ++j) {
#pragma unroll
      for (int r = 0; r < 8; ++r) {
        int m = m0 + 16 * i + r + 8 * half;
        int n = n0 + 16 * j + lc;
        float vv = acc[i][j][r] + bias[n];
        int bb = m >> 11, t = m & (TB - 1);
        int h  = n >> 6,  dd = n & 63;
        if (OUTMODE == 0)
          out[(((size_t)bb * HB + h) * TB + t) * DB + dd] = (__bf16)vv;
        else
          out[(((size_t)bb * HB + h) * DB + dd) * TB + t] = (__bf16)vv;
      }
    }
  }
}

// ---------------- register-blocked GEMM: 32x64 per wave, f32 row-major output ----------------
__global__ void k_gemm_rowout(const __bf16* __restrict__ A, const __bf16* __restrict__ Bt,
                              const float* __restrict__ bias, float* __restrict__ out) {
  const int K = CB;
  int w    = blockIdx.x * (blockDim.x >> 5) + (threadIdx.x >> 5);
  int lane = threadIdx.x & 31, half = lane >> 4, lc = lane & 15;
  int mtile = w >> 4, ntile = w & 15;
  int m0 = mtile << 5, n0 = ntile << 6;

  const __bf16* arow0 = A + (size_t)(m0 + lc) * K;
  const __bf16* arow1 = A + (size_t)(m0 + 16 + lc) * K;
  const __bf16* brow[4];
#pragma unroll
  for (int j = 0; j < 4; ++j)
    brow[j] = Bt + (size_t)(n0 + 16 * j + lc) * K + half * 16;

  v8f acc[2][4] = {};
  for (int kk = 0; kk < K; kk += 32) {
    v16bf a0 = cat8(*(const v8bf*)(arow0 + kk + half * 16),
                    *(const v8bf*)(arow0 + kk + 8 + half * 16));
    v16bf a1 = cat8(*(const v8bf*)(arow1 + kk + half * 16),
                    *(const v8bf*)(arow1 + kk + 8 + half * 16));
    v16bf b[4];
#pragma unroll
    for (int j = 0; j < 4; ++j) b[j] = *(const v16bf*)(brow[j] + kk);
#pragma unroll
    for (int j = 0; j < 4; ++j) {
      acc[0][j] = wmma_bf16(a0, b[j], acc[0][j]);
      acc[1][j] = wmma_bf16(a1, b[j], acc[1][j]);
    }
  }
#pragma unroll
  for (int i = 0; i < 2; ++i) {
#pragma unroll
    for (int j = 0; j < 4; ++j) {
#pragma unroll
      for (int r = 0; r < 8; ++r) {
        int m = m0 + 16 * i + r + 8 * half;
        int n = n0 + 16 * j + lc;
        out[(size_t)m * CB + n] = acc[i][j][r] + bias[n];
      }
    }
  }
}

// ---------------- RoPE in place on bf16 [B,H,T,D] ----------------
__global__ void k_rope(__bf16* __restrict__ t_, int total_pairs) {
  int i = blockIdx.x * blockDim.x + threadIdx.x;
  if (i >= total_pairs) return;
  int dd2 = i & 31;                  // pair index within head dim (D/2 = 32)
  int t   = (i >> 5) & (TB - 1);
  int bh  = i >> 16;                 // 32*2048 = 65536
  size_t base = (((size_t)bh * TB) + t) * DB + dd2 * 2;
  float x0 = (float)t_[base], x1 = (float)t_[base + 1];
  float theta = __powf(10000.f, -(float)(2 * dd2) / 64.f);
  float ang = (float)t * theta;
  float s, c;
  __sincosf(ang, &s, &c);
  t_[base]     = (__bf16)(x0 * c - x1 * s);
  t_[base + 1] = (__bf16)(x1 * c + x0 * s);
}

// ---------------- flash attention ----------------
// block = 128 threads = 4 waves, each wave owns a 16-row query tile; 32 keys/iter.
// q,k: [B,H,T,D] bf16 (RoPE'd); v: [B,H,D,T] bf16 (transposed -> contiguous B-frags)
__global__ void k_flash(const __bf16* __restrict__ q, const __bf16* __restrict__ kmat,
                        const __bf16* __restrict__ vmat, const int* __restrict__ pad,
                        __bf16* __restrict__ yatt) {
  __shared__ __bf16 plds[4][16][32];       // per-wave P staging (C-layout -> A-layout)
  int wave = threadIdx.x >> 5;
  int lane = threadIdx.x & 31, half = lane >> 4, lc = lane & 15;
  int bh = blockIdx.y;
  int b_ = bh >> 4, h = bh & 15;
  int qb = blockIdx.x * 64 + wave * 16;
  const __bf16* Qb = q    + (size_t)bh * TB * DB;
  const __bf16* Kb = kmat + (size_t)bh * TB * DB;
  const __bf16* Vt = vmat + (size_t)bh * DB * TB;   // [D][T]
  const int* padb = pad + b_ * TB;

  // preload Q fragment (16x64 = two 16x32 A fragments)
  const __bf16* qrowp = Qb + (size_t)(qb + lc) * DB;
  v16bf aq[2];
#pragma unroll
  for (int c = 0; c < 2; ++c)
    aq[c] = cat8(*(const v8bf*)(qrowp + c * 32 + half * 16),
                 *(const v8bf*)(qrowp + c * 32 + 8 + half * 16));

  float mrow[8], lrow[8];
  v8f o[4] = {};
#pragma unroll
  for (int r = 0; r < 8; ++r) { mrow[r] = -1e30f; lrow[r] = 0.f; }

  int kend = qb + 16;                       // causal: keys 0..qb+15 needed
  for (int kb = 0; kb < kend; kb += 32) {
    v8f s0 = {}, s1 = {};
#pragma unroll
    for (int c = 0; c < 2; ++c) {
      v16bf bk0 = *(const v16bf*)(Kb + (size_t)(kb + lc) * DB + c * 32 + half * 16);
      v16bf bk1 = *(const v16bf*)(Kb + (size_t)(kb + 16 + lc) * DB + c * 32 + half * 16);
      s0 = wmma_bf16(aq[c], bk0, s0);
      s1 = wmma_bf16(aq[c], bk1, s1);
    }
    int key0 = kb + lc, key1 = kb + 16 + lc;
    bool ok0 = padb[key0] != 0;
    bool ok1 = padb[key1] != 0;
#pragma unroll
    for (int r = 0; r < 8; ++r) {
      int qrow = qb + r + 8 * half;
      float v0 = (ok0 && key0 <= qrow) ? s0[r] * 0.125f : -1e30f;   // 1/sqrt(64)
      float v1 = (ok1 && key1 <= qrow) ? s1[r] * 0.125f : -1e30f;
      float mx = fmaxf(v0, v1);
#pragma unroll
      for (int off = 8; off >= 1; off >>= 1) mx = fmaxf(mx, __shfl_xor(mx, off, 16));
      float mnew  = fmaxf(mrow[r], mx);
      float alpha = __expf(mrow[r] - mnew);
      float p0 = (v0 > -1e29f) ? __expf(v0 - mnew) : 0.f;
      float p1 = (v1 > -1e29f) ? __expf(v1 - mnew) : 0.f;
      float rs = p0 + p1;
#pragma unroll
      for (int off = 8; off >= 1; off >>= 1) rs += __shfl_xor(rs, off, 16);
      lrow[r] = lrow[r] * alpha + rs;
      mrow[r] = mnew;
#pragma unroll
      for (int nt = 0; nt < 4; ++nt) o[nt][r] *= alpha;
      plds[wave][r + 8 * half][lc]      = (__bf16)p0;
      plds[wave][r + 8 * half][16 + lc] = (__bf16)p1;
    }
    // same-wave LDS read-back: DS ops are in-order within a wave
    v16bf pa = cat8(*(const v8bf*)&plds[wave][lc][half * 16],
                    *(const v8bf*)&plds[wave][lc][8 + half * 16]);
#pragma unroll
    for (int nt = 0; nt < 4; ++nt) {
      // B(key, dim): lane holds dim = nt*16+lc, elements j -> key = kb + half*16 + j
      v16bf bv = *(const v16bf*)(Vt + (size_t)(nt * 16 + lc) * TB + kb + half * 16);
      o[nt] = wmma_bf16(pa, bv, o[nt]);
    }
  }
  // epilogue: normalize and write [B,T,H*D] bf16 row-major
#pragma unroll
  for (int r = 0; r < 8; ++r) {
    float inv = lrow[r] > 0.f ? 1.f / lrow[r] : 0.f;
    int qrow = qb + r + 8 * half;
#pragma unroll
    for (int nt = 0; nt < 4; ++nt) {
      int dd = nt * 16 + lc;
      yatt[((size_t)b_ * TB + qrow) * CB + h * DB + dd] = (__bf16)(o[nt][r] * inv);
    }
  }
}

// ---------------- LayerNorm over last dim (1024) ----------------
__global__ void k_lnorm(const float* __restrict__ y, const float* __restrict__ gamma,
                        const float* __restrict__ beta, float* __restrict__ out) {
  __shared__ float red[256];
  int m = blockIdx.x;
  const float* row = y + (size_t)m * CB;
  float s = 0.f;
  for (int i = threadIdx.x; i < CB; i += 256) s += row[i];
  red[threadIdx.x] = s;
  __syncthreads();
  for (int st = 128; st > 0; st >>= 1) {
    if (threadIdx.x < st) red[threadIdx.x] += red[threadIdx.x + st];
    __syncthreads();
  }
  float mu = red[0] * (1.f / CB);
  __syncthreads();
  float v = 0.f;
  for (int i = threadIdx.x; i < CB; i += 256) { float d = row[i] - mu; v += d * d; }
  red[threadIdx.x] = v;
  __syncthreads();
  for (int st = 128; st > 0; st >>= 1) {
    if (threadIdx.x < st) red[threadIdx.x] += red[threadIdx.x + st];
    __syncthreads();
  }
  float rstd = rsqrtf(red[0] * (1.f / CB) + 1e-5f);
  for (int i = threadIdx.x; i < CB; i += 256)
    out[(size_t)m * CB + i] = (row[i] - mu) * rstd * gamma[i] + beta[i];
}

// ---------------- host launch ----------------
extern "C" void kernel_launch(void* const* d_in, const int* in_sizes, int n_in,
                              void* d_out, int out_size, void* d_ws, size_t ws_size,
                              hipStream_t stream) {
  const float* x     = (const float*)d_in[0];
  const float* Wq    = (const float*)d_in[1];
  const float* bq    = (const float*)d_in[2];
  const float* Wk    = (const float*)d_in[3];
  const float* bk    = (const float*)d_in[4];
  const float* Wv    = (const float*)d_in[5];
  const float* bv    = (const float*)d_in[6];
  const float* Wo    = (const float*)d_in[7];
  const float* bo    = (const float*)d_in[8];
  const float* gamma = (const float*)d_in[9];
  const float* beta  = (const float*)d_in[10];
  const int*   pad   = (const int*)d_in[11];

  char* ws = (char*)d_ws;
  const size_t MB = 1ull << 20;
  __bf16* xb   = (__bf16*)(ws + 0 * MB);    // 8 MiB  [4096,1024] bf16
  __bf16* WqT  = (__bf16*)(ws + 8 * MB);    // 2 MiB  [N,K] bf16
  __bf16* WkT  = (__bf16*)(ws + 10 * MB);
  __bf16* WvT  = (__bf16*)(ws + 12 * MB);
  __bf16* WoT  = (__bf16*)(ws + 14 * MB);
  __bf16* qh   = (__bf16*)(ws + 16 * MB);   // 8 MiB  [B,H,T,D]
  __bf16* kh   = (__bf16*)(ws + 24 * MB);
  __bf16* vh   = (__bf16*)(ws + 32 * MB);   // 8 MiB  [B,H,D,T] (transposed)
  __bf16* yatt = (__bf16*)(ws + 40 * MB);   // 8 MiB  [B*T, C]
  float*  y2   = (float*)(ws + 48 * MB);    // 16 MiB [B*T, C] f32

  const int NX = 2 * 2048 * 1024;           // x elements
  k_cvt_bf16<<<NX / 256, 256, 0, stream>>>(x, xb, NX);
  k_transpose_bf16<<<(1024 * 1024) / 256, 256, 0, stream>>>(Wq, WqT);
  k_transpose_bf16<<<(1024 * 1024) / 256, 256, 0, stream>>>(Wk, WkT);
  k_transpose_bf16<<<(1024 * 1024) / 256, 256, 0, stream>>>(Wv, WvT);
  k_transpose_bf16<<<(1024 * 1024) / 256, 256, 0, stream>>>(Wo, WoT);

  // QKV projections: 2048 wave-tiles (32x64 each), 4 waves/block -> 512 blocks
  k_gemm_bf16out<0><<<512, 128, 0, stream>>>(xb, WqT, bq, qh);
  k_gemm_bf16out<0><<<512, 128, 0, stream>>>(xb, WkT, bk, kh);
  k_gemm_bf16out<1><<<512, 128, 0, stream>>>(xb, WvT, bv, vh);

  const int PAIRS = 2 * 16 * 2048 * 32;     // 2,097,152 rotation pairs
  k_rope<<<PAIRS / 256, 256, 0, stream>>>(qh, PAIRS);
  k_rope<<<PAIRS / 256, 256, 0, stream>>>(kh, PAIRS);

  dim3 fg(2048 / 64, 2 * 16);               // (q-blocks of 64 rows, B*H)
  k_flash<<<fg, 128, 0, stream>>>(qh, kh, vh, pad, yatt);

  k_gemm_rowout<<<512, 128, 0, stream>>>(yatt, WoT, bo, y2);
  k_lnorm<<<4096, 256, 0, stream>>>(y2, gamma, beta, (float*)d_out);
}
